// Attention_57853209477443
// MI455X (gfx1250) — matrix operand order repaired and checked by reading, not compiled, not run on hardware
//
#include <hip/hip_runtime.h>
#include <hip/hip_bf16.h>

// ---- problem constants (match reference) ----
#define NB   2          // batch
#define NSEQ 2048       // sequence length N
#define NCQ  256        // input channels
#define NCH  256        // qkv channels
#define NH   8          // heads
#define ND   32         // head dim
#define NBN  (NB*NSEQ)  // 4096 flattened rows

typedef _Float16 half_t;
typedef __attribute__((ext_vector_type(16))) _Float16 v16h;
typedef __attribute__((ext_vector_type(2)))  _Float16 v2h;
typedef __attribute__((ext_vector_type(8)))  float    v8f;

// ---------------- WMMA helpers ----------------

static __device__ __forceinline__ v8f wmma16(v16h a, v16h b, v8f c) {
    // D = A(16x32 f16) * B(32x16 f16) + C(16x16 f32)
    return __builtin_amdgcn_wmma_f32_16x16x32_f16(
        /*neg_a=*/false, a, /*neg_b=*/false, b,
        /*c_mod=*/(short)0, c, /*reuse_a=*/false, /*reuse_b=*/false);
}

static __device__ __forceinline__ float fast_rcp(float x) {
    return __builtin_amdgcn_rcpf(x);       // v_rcp_f32 (enough for f16 outputs)
}

// A-fragment as packed by this code (16x32 f16, M x K, contiguous k): lane<16 holds row M=lane, K pairs
//   VGPR v -> K=2v,2v+1 (K=0..15); upper half-wave +16 (K=16..31).
static __device__ __forceinline__ int a_kpair(int i, int hi) {
    return 2 * i + hi * 16;
}

// Load A fragment from row-major f32 source (convert to f16).
static __device__ __forceinline__ v16h load_a_f32(const float* src, int ld,
                                                  int m, int k0, int hi) {
    v16h a;
    const float* row = src + (size_t)m * ld + k0;
#pragma unroll
    for (int i = 0; i < 8; ++i) {
        int kp = a_kpair(i, hi);
        a[2 * i]     = (_Float16)row[kp];
        a[2 * i + 1] = (_Float16)row[kp + 1];
    }
    return a;
}

// Load A fragment from row-major f16 source (pairs are contiguous -> v2h loads).
static __device__ __forceinline__ v16h load_a_f16(const half_t* src, int ld,
                                                  int m, int k0, int hi) {
    v16h a;
    const half_t* row = src + (size_t)m * ld + k0;
#pragma unroll
    for (int i = 0; i < 8; ++i) {
        v2h p = *(const v2h*)(row + a_kpair(i, hi));
        a[2 * i] = p[0]; a[2 * i + 1] = p[1];
    }
    return a;
}

// B-fragment (32x16 f16, K x N): lane<16 -> col N=lane&15, K=0..15 (VGPR v: K=2v,2v+1);
// lanes 16..31 -> same col, K=16..31.  (Same contiguous k order as the A packing, not the ISA table order.)

// Load B from row-major f32 weight W[K][ld] (stride ld between K rows).
static __device__ __forceinline__ v16h load_b_w(const float* W, int ld,
                                                int k0, int n, int hi) {
    v16h b;
#pragma unroll
    for (int i = 0; i < 8; ++i) {
        int k = k0 + hi * 16 + 2 * i;
        b[2 * i]     = (_Float16)W[(size_t)k * ld + n];
        b[2 * i + 1] = (_Float16)W[(size_t)(k + 1) * ld + n];
    }
    return b;
}

// Load B = K^T tile for scores: B[d][kpos] = kh[(kbase+kpos)*NCH + doff + d]
// K-dim (d) is contiguous in memory -> v2h loads.
static __device__ __forceinline__ v16h load_b_kT(const half_t* kh, int kbase,
                                                 int n /*kpos col*/, int doff, int hi) {
    v16h b;
    const half_t* row = kh + (size_t)(kbase + n) * NCH + doff + hi * 16;
#pragma unroll
    for (int i = 0; i < 8; ++i) {
        v2h p = *(const v2h*)(row + 2 * i);
        b[2 * i] = p[0]; b[2 * i + 1] = p[1];
    }
    return b;
}

// Load B = V tile: B[kpos][d] = vh[(k0+kpos)*NCH + doff + d]; K-dim strided by NCH.
static __device__ __forceinline__ v16h load_b_v(const half_t* vh, int k0,
                                                int n /*d col*/, int doff, int hi) {
    v16h b;
#pragma unroll
    for (int i = 0; i < 8; ++i) {
        int kk = k0 + hi * 16 + 2 * i;
        b[2 * i]     = vh[(size_t)kk * NCH + doff + n];
        b[2 * i + 1] = vh[(size_t)(kk + 1) * NCH + doff + n];
    }
    return b;
}

// ---------------- Kernel 1: fused Q/K/V/Gate projections ----------------
// grid = (NBN/16, 32); blockIdx.y>>3 selects {Wq,Wk,Wv,Wg}; one wave per 16x32 tile
// (two 16x16 WMMA tiles -> A fragment loaded once per K-step, used twice).
__global__ void proj_kernel(const float* __restrict__ qx,
                            const float* __restrict__ Wq, const float* __restrict__ bq,
                            const float* __restrict__ Wk, const float* __restrict__ Wv,
                            const float* __restrict__ Wg, const float* __restrict__ bg,
                            const float* __restrict__ gbias,
                            half_t* __restrict__ qh, half_t* __restrict__ kh,
                            half_t* __restrict__ vh, float* __restrict__ gate) {
    const int lane = threadIdx.x & 31;
    const int hi   = lane >> 4;
    const int col  = lane & 15;
    const int m0   = blockIdx.x * 16;
    const int w    = blockIdx.y >> 3;
    const int n0   = (blockIdx.y & 7) * 32;
    const float* W = (w == 0) ? Wq : (w == 1) ? Wk : (w == 2) ? Wv : Wg;

    v8f acc0 = {}, acc1 = {};
    for (int k0 = 0; k0 < NCQ; k0 += 32) {
        v16h a  = load_a_f32(qx, NCQ, m0 + col, k0, hi);
        v16h b0 = load_b_w(W, NCH, k0, n0 + col,      hi);
        v16h b1 = load_b_w(W, NCH, k0, n0 + 16 + col, hi);
        acc0 = wmma16(a, b0, acc0);
        acc1 = wmma16(a, b1, acc1);
    }

#pragma unroll
    for (int half_n = 0; half_n < 2; ++half_n) {
        const int n = n0 + half_n * 16 + col;
#pragma unroll
        for (int r = 0; r < 8; ++r) {
            const int m = m0 + r + hi * 8;   // C-layout: vgpr r, upper half-wave = rows 8..15
            float v = half_n ? acc1[r] : acc0[r];
            if (w == 0) {                     // q: (+bq) * 1/sqrt(D)
                v = (v + bq[n]) * 0.17677669529663687f;
                qh[(size_t)m * NCH + n] = (half_t)v;
            } else if (w == 1) {
                kh[(size_t)m * NCH + n] = (half_t)v;
            } else if (w == 2) {
                vh[(size_t)m * NCH + n] = (half_t)v;
            } else {                          // gate = sigmoid(x + bg + gbias), fast rcp
                v = v + bg[n] + gbias[n];
                gate[(size_t)m * NCQ + n] = fast_rcp(1.0f + __expf(-v));
            }
        }
    }
}

// ---------------- Kernel 2: flash attention with pair bias ----------------
// grid = (NSEQ/16, NH, NB); one wave handles a 16-row q-tile for one (b,h).
// k-chunk = 64 (4 score tiles / 8 WMMA per iter) to amortize shuffle trees & exp.
// The 268MB bias tensor is streamed ONCE and exceeds the 192MB L2 -> load it
// non-temporal (TH=NT) so it cannot evict the heavily-reused k/v f16 buffers.
__global__ void attn_kernel(const half_t* __restrict__ qh,
                            const half_t* __restrict__ kh,
                            const half_t* __restrict__ vh,
                            const float*  __restrict__ bias,
                            half_t* __restrict__ oh) {
    __shared__ _Float16 Pt[16][66];        // 16x64 P tile (+pad), one wave per block

    const int lane = threadIdx.x & 31;
    const int hi   = lane >> 4;
    const int col  = lane & 15;
    const int q0   = blockIdx.x * 16;
    const int h    = blockIdx.y;
    const int b    = blockIdx.z;
    const int doff = h * ND;
    const int kvbase = b * NSEQ;           // rows of this batch in the flat [NBN,NCH] buffers

    // q A-fragment is reused across the whole loop
    const v16h qa = load_a_f16(qh, NCH, kvbase + q0 + col, doff, hi);

    v8f O0 = {}, O1 = {};
    float mrow[8], lrow[8];
#pragma unroll
    for (int r = 0; r < 8; ++r) { mrow[r] = -1e30f; lrow[r] = 0.0f; }

    const float* bias_base = bias + (((size_t)b * NH + h) * NSEQ + q0) * NSEQ;

    for (int k0 = 0; k0 < NSEQ; k0 += 64) {
        // prefetch next bias chunk (global_prefetch_b8)
        if (k0 + 64 < NSEQ)
            __builtin_prefetch(bias_base + (size_t)(hi * 8) * NSEQ + k0 + 64, 0, 0);

        // S = q @ k^T : four 16x16 tiles covering kpos [k0, k0+64)
        v8f z = {};
        v8f s0, s1, s2, s3;
        {
            v16h kb0 = load_b_kT(kh, kvbase + k0,      col, doff, hi);
            v16h kb1 = load_b_kT(kh, kvbase + k0 + 16, col, doff, hi);
            v16h kb2 = load_b_kT(kh, kvbase + k0 + 32, col, doff, hi);
            v16h kb3 = load_b_kT(kh, kvbase + k0 + 48, col, doff, hi);
            s0 = wmma16(qa, kb0, z);
            s1 = wmma16(qa, kb1, z);
            s2 = wmma16(qa, kb2, z);
            s3 = wmma16(qa, kb3, z);
        }

        // add pair bias: non-temporal stream (read-once, > L2 capacity);
        // coalesced: 16 lanes read 16 consecutive floats per row.
#pragma unroll
        for (int r = 0; r < 8; ++r) {
            const float* br = bias_base + (size_t)(r + hi * 8) * NSEQ + k0;
            s0[r] += __builtin_nontemporal_load(br + col);
            s1[r] += __builtin_nontemporal_load(br + 16 + col);
            s2[r] += __builtin_nontemporal_load(br + 32 + col);
            s3[r] += __builtin_nontemporal_load(br + 48 + col);
        }

        // online softmax (row r lives across the 16 lanes of this half-wave)
#pragma unroll
        for (int r = 0; r < 8; ++r) {
            float x = fmaxf(fmaxf(s0[r], s1[r]), fmaxf(s2[r], s3[r]));
#pragma unroll
            for (int off = 1; off < 16; off <<= 1)
                x = fmaxf(x, __shfl_xor(x, off, 32));
            const float mnew  = fmaxf(mrow[r], x);
            const float scale = __expf(mrow[r] - mnew);
            mrow[r] = mnew;
            const float p0 = __expf(s0[r] - mnew);
            const float p1 = __expf(s1[r] - mnew);
            const float p2 = __expf(s2[r] - mnew);
            const float p3 = __expf(s3[r] - mnew);
            float ssum = (p0 + p1) + (p2 + p3);
#pragma unroll
            for (int off = 1; off < 16; off <<= 1)
                ssum += __shfl_xor(ssum, off, 32);
            lrow[r] = lrow[r] * scale + ssum;
            O0[r] *= scale;
            O1[r] *= scale;
            const int m = r + hi * 8;
            Pt[m][col]      = (_Float16)p0;   // stage P for C->A layout reshape
            Pt[m][16 + col] = (_Float16)p1;
            Pt[m][32 + col] = (_Float16)p2;
            Pt[m][48 + col] = (_Float16)p3;
        }
        __asm volatile("s_wait_dscnt 0" ::: "memory");

        // reload P as two A-fragments (16 q-rows x 32 kpos each)
        v16h pa0, pa1;
#pragma unroll
        for (int i = 0; i < 8; ++i) {
            const int kp = a_kpair(i, hi);
            pa0[2 * i]     = Pt[col][kp];
            pa0[2 * i + 1] = Pt[col][kp + 1];
            pa1[2 * i]     = Pt[col][32 + kp];
            pa1[2 * i + 1] = Pt[col][32 + kp + 1];
        }

        // O += P @ V  (two d-halves x two kpos-halves)
        v16h vb00 = load_b_v(vh, kvbase + k0,      col, doff,      hi);
        v16h vb01 = load_b_v(vh, kvbase + k0,      col, doff + 16, hi);
        v16h vb10 = load_b_v(vh, kvbase + k0 + 32, col, doff,      hi);
        v16h vb11 = load_b_v(vh, kvbase + k0 + 32, col, doff + 16, hi);
        O0 = wmma16(pa0, vb00, O0);
        O1 = wmma16(pa0, vb01, O1);
        O0 = wmma16(pa1, vb10, O0);
        O1 = wmma16(pa1, vb11, O1);
        // per-wave LDS ops are in-order: next iteration's stores cannot pass these loads
    }

    // normalize and store o (merged-head layout [NBN, NCH], f16)
#pragma unroll
    for (int r = 0; r < 8; ++r) {
        const float inv = fast_rcp(lrow[r]);
        const int m = kvbase + q0 + r + hi * 8;
        oh[(size_t)m * NCH + doff + col]      = (half_t)(O0[r] * inv);
        oh[(size_t)m * NCH + doff + 16 + col] = (half_t)(O1[r] * inv);
    }
}

// ---------------- Kernel 3: output projection * gate ----------------
// grid = (NBN/16, NCQ/32); one wave per 16x32 output tile (A loaded once, 2 WMMA).
__global__ void out_kernel(const half_t* __restrict__ oh,
                           const float* __restrict__ Wout,
                           const float* __restrict__ bout,
                           const float* __restrict__ gate,
                           float* __restrict__ out) {
    const int lane = threadIdx.x & 31;
    const int hi   = lane >> 4;
    const int col  = lane & 15;
    const int m0   = blockIdx.x * 16;
    const int n0   = blockIdx.y * 32;

    v8f acc0 = {}, acc1 = {};
    for (int k0 = 0; k0 < NCH; k0 += 32) {
        v16h a  = load_a_f16(oh, NCH, m0 + col, k0, hi);
        v16h b0 = load_b_w(Wout, NCQ, k0, n0 + col,      hi);
        v16h b1 = load_b_w(Wout, NCQ, k0, n0 + 16 + col, hi);
        acc0 = wmma16(a, b0, acc0);
        acc1 = wmma16(a, b1, acc1);
    }

#pragma unroll
    for (int half_n = 0; half_n < 2; ++half_n) {
        const int n = n0 + half_n * 16 + col;
#pragma unroll
        for (int r = 0; r < 8; ++r) {
            const int m = m0 + r + hi * 8;
            const float v = (half_n ? acc1[r] : acc0[r]) + bout[n];
            // write-once result: non-temporal store, don't pollute L2
            __builtin_nontemporal_store(gate[(size_t)m * NCQ + n] * v,
                                        out + (size_t)m * NCQ + n);
        }
    }
}

// ---------------- launch ----------------
extern "C" void kernel_launch(void* const* d_in, const int* in_sizes, int n_in,
                              void* d_out, int out_size, void* d_ws, size_t ws_size,
                              hipStream_t stream) {
    const float* q_x   = (const float*)d_in[0];
    const float* abias = (const float*)d_in[1];
    const float* Wq    = (const float*)d_in[2];
    const float* bq    = (const float*)d_in[3];
    const float* Wk    = (const float*)d_in[4];
    const float* Wv    = (const float*)d_in[5];
    const float* Wout  = (const float*)d_in[6];
    const float* bout  = (const float*)d_in[7];
    const float* Wg    = (const float*)d_in[8];
    const float* bg    = (const float*)d_in[9];
    const float* gbias = (const float*)d_in[10];
    float* out = (float*)d_out;

    // workspace layout (bytes): q 2MB | k 2MB | v 2MB | o 2MB | gate 4MB
    char* ws = (char*)d_ws;
    const size_t seg = (size_t)NBN * NCH * sizeof(half_t);   // 2 MiB
    half_t* qh   = (half_t*)(ws);
    half_t* kh   = (half_t*)(ws + seg);
    half_t* vh   = (half_t*)(ws + 2 * seg);
    half_t* oh   = (half_t*)(ws + 3 * seg);
    float*  gate = (float*)(ws + 4 * seg);

    proj_kernel<<<dim3(NBN / 16, 32), 32, 0, stream>>>(
        q_x, Wq, bq, Wk, Wv, Wg, bg, gbias, qh, kh, vh, gate);

    attn_kernel<<<dim3(NSEQ / 16, NH, NB), 32, 0, stream>>>(
        qh, kh, vh, abias, oh);

    out_kernel<<<dim3(NBN / 16, NCQ / 32), 32, 0, stream>>>(
        oh, Wout, bout, gate, out);
}
